// Decoder_2637109919852
// MI455X (gfx1250) — matrix operand order repaired and checked by reading, not compiled, not run on hardware
//
#include <hip/hip_runtime.h>

typedef _Float16 half_t;
typedef __attribute__((ext_vector_type(16))) _Float16 v16h;
typedef __attribute__((ext_vector_type(8)))  _Float16 v8h;
typedef __attribute__((ext_vector_type(8)))  float    v8f;

#define DEVINL __device__ __forceinline__

DEVINL float fsigm(float x) { return 1.0f / (1.0f + __expf(-x)); }
DEVINL float ftanh(float x)
{
    float e = __expf(-2.0f * fabsf(x));       // in (0,1], no overflow
    float r = (1.0f - e) / (1.0f + e);
    return copysignf(r, x);
}

// ---------------------------------------------------------------------------
// WMMA tile: C(16x16) += A(16x32 f16, LDS) * B(32x16 f16, pre-swizzled global)
// A layout (ISA 7.12.2, 16-bit A 16x32): lane l holds row M=l&15; halves 0..7
// are K = k0 + (l<16?0:8) + i, halves 8..15 are K = +16.  -> two 16B ds loads.
// B swizzled so each lane's 16 halves are one contiguous 32B global load.
// ---------------------------------------------------------------------------
DEVINL v8f wmma_tile(const half_t* __restrict__ sA, int Kp, int mbase,
                     const half_t* __restrict__ Bswz, int NT, int nt, int KT,
                     float binit, int lane)
{
    const int kAdd = (lane & 16) ? 8 : 0;
    const int m    = mbase + (lane & 15);
    v8f acc;
#pragma unroll
    for (int r = 0; r < 8; ++r) acc[r] = binit;
    for (int kt = 0; kt < KT; ++kt) {
        const half_t* ar = sA + m * Kp + kt * 32 + kAdd;
        v8h lo = *(const v8h*)(ar);
        v8h hi = *(const v8h*)(ar + 16);
        v16h a;
#pragma unroll
        for (int i = 0; i < 8; ++i) { a[i] = lo[i]; a[8 + i] = hi[i]; }
        v16h b = *(const v16h*)(Bswz + (((size_t)kt * NT + nt) << 9) + (lane << 4));
        acc = __builtin_amdgcn_wmma_f32_16x16x32_f16(false, a, false, b,
                                                     (short)0, acc, false, false);
    }
    return acc;
}

// GEMM with one 16-row M-tile (rows 8..15 may be zero padding).
DEVINL void gemm16(const half_t* __restrict__ sA, int Kp,
                   const half_t* __restrict__ Bswz, int NT, int KT,
                   const float* __restrict__ bias,
                   float* __restrict__ outF, int outStride, int lane, int wave)
{
    const int mAdd = (lane & 16) ? 8 : 0;
    for (int nt = wave; nt < NT; nt += 16) {
        int n = nt * 16 + (lane & 15);
        float bv = bias ? bias[n] : 0.0f;
        v8f acc = wmma_tile(sA, Kp, 0, Bswz, NT, nt, KT, bv, lane);
#pragma unroll
        for (int r = 0; r < 8; ++r) outF[(r + mAdd) * outStride + n] = acc[r];
    }
}

// ---------------------------------------------------------------------------
// Weight swizzle: W (N x K, row-major f32) -> B-fragment order f16.
// ---------------------------------------------------------------------------
__global__ void swz_kernel(const float* __restrict__ W, half_t* __restrict__ dst,
                           int K, int N, int Kp, int Np)
{
    int gid = blockIdx.x * blockDim.x + threadIdx.x;
    int total = Kp * Np;
    if (gid >= total) return;
    int k = gid / Np, n = gid - k * Np;
    float v = (k < K && n < N) ? W[(size_t)n * K + k] : 0.0f;
    int kt = k >> 5, ki = k & 31, nt = n >> 4;
    int lanesel = (((ki >> 3) & 1) ? 16 : 0) + (n & 15);
    int i = (ki & 7) + ((ki & 16) ? 8 : 0);
    size_t off = (((size_t)kt * (Np >> 4) + nt) << 9) + ((size_t)lanesel << 4) + i;
    dst[off] = (half_t)v;
}

// ---------------------------------------------------------------------------
// proc_annot = inputs @ a_W.T + a_b  (32768 x 128, K=256), f16 output (time
// invariant across the 200 decoder steps -> precompute once).
// ---------------------------------------------------------------------------
__global__ void __launch_bounds__(256, 4)
annot_kernel(const float* __restrict__ inputs, const half_t* __restrict__ aWs,
             const float* __restrict__ a_b, half_t* __restrict__ annot)
{
    const int lane = threadIdx.x & 31;
    const int wave = threadIdx.x >> 5;
    int gt = blockIdx.x * 8 + wave;          // 16384 tiles = 2048 mtiles x 8 ntiles
    int mt = gt >> 3, nt = gt & 7;
    const int kAdd = (lane & 16) ? 8 : 0;
    const int mAdd = (lane & 16) ? 8 : 0;
    int row = mt * 16 + (lane & 15);
    int n   = nt * 16 + (lane & 15);
    float bv = a_b[n];
    v8f acc;
#pragma unroll
    for (int r = 0; r < 8; ++r) acc[r] = bv;
    for (int kt = 0; kt < 8; ++kt) {
        const float* ap = inputs + (size_t)row * 256 + kt * 32 + kAdd;
        v16h a;
#pragma unroll
        for (int i = 0; i < 16; ++i) a[i] = (half_t)ap[i + ((i >= 8) ? 8 : 0)];
        v16h b = *(const v16h*)(aWs + (((size_t)kt * 8 + nt) << 9) + (lane << 4));
        acc = __builtin_amdgcn_wmma_f32_16x16x32_f16(false, a, false, b,
                                                     (short)0, acc, false, false);
    }
#pragma unroll
    for (int r = 0; r < 8; ++r)
        annot[((size_t)mt * 16 + r + mAdd) * 128 + n] = (half_t)acc[r];
}

// ---------------------------------------------------------------------------
// Persistent decoder: 1 WG = 8 batch rows, 200 timesteps, all recurrent state
// (h_a, h1, h2, ctx, attn, attn_cum) resident in LDS for the whole sequence.
// Attention over T=512 processed in 8 chunks of 64 -> 512-row WMMA tiles.
// ---------------------------------------------------------------------------
struct MainArgs {
    const float* inputs; const float* memory; const unsigned char* mask;
    const float *pre_b1, *pre_b2, *arnn_bih, *arnn_bhh, *loc_b, *q_b, *v_W;
    const float *proj_b, *d1_bih, *d1_bhh, *d2_bih, *d2_bhh;
    const float *mel_b, *stop_W, *stop_b, *attn_init, *mem_init, *dec_init;
    const half_t* annot;
    const half_t *preW1s, *preW2s, *aWihs, *aWhhs, *qWs, *locWs, *convS,
                 *projWs, *d1ihs, *d1hhs, *d2ihs, *d2hhs, *melWs;
    float* outMel; float* outAlign; float* outStop;
};

// LDS layout (bytes)
#define L_SX     0        // 16*512 f16 staging (rows 8..15 zero)      16384
#define L_SG     16384    // 2*16*768 f32 gi/gh  (union: im2col+locf)  98304
#define L_SE     114688   // 8*512 f32 energy/align                    16384
#define L_SAQ    131072   // 16*128 f32 proc_q                          8192
#define L_SHA    139264   // 8*256 f32                                  8192
#define L_SH1    147456
#define L_SH2    155648
#define L_SCTX   163840
#define L_SD     172032   // 16*256 f32 dec buffer                     16384
#define L_SATTN  188416   // 8*512 f32 attention (recurrent)           16384
#define L_SCUM   204800   // 8*512 f32 cumulative attention            16384
#define L_SMASK  221184   // 8*512 u8                                   4096
#define L_SSW    225280   // 656 f32 stop_W                             2688
#define L_SV     227968   // 128 f32 v_W                                 512
#define L_SRS    228480   // 16 f32
#define L_SSTOP  228544   // 16 f32
#define LDS_BYTES 228608

__global__ void __launch_bounds__(512, 1) decoder_main(MainArgs A)
{
    extern __shared__ char smem[];
    half_t* sX   = (half_t*)(smem + L_SX);
    float*  sG0  = (float*) (smem + L_SG);
    float*  sG1  = sG0 + 16 * 768;
    half_t* sIm  = (half_t*)(smem + L_SG);            // 512 rows x 64 K (f16)
    half_t* sLoc = (half_t*)(smem + L_SG + 65536);    // 512 rows x 32 N (f16)
    float*  sE   = (float*) (smem + L_SE);
    float*  sAq  = (float*) (smem + L_SAQ);
    float*  sHa  = (float*) (smem + L_SHA);
    float*  sH1  = (float*) (smem + L_SH1);
    float*  sH2  = (float*) (smem + L_SH2);
    float*  sCtx = (float*) (smem + L_SCTX);
    float*  sD   = (float*) (smem + L_SD);
    float*  sAt  = (float*) (smem + L_SATTN);
    float*  sCu  = (float*) (smem + L_SCUM);
    unsigned char* sMask = (unsigned char*)(smem + L_SMASK);
    float*  sSW  = (float*) (smem + L_SSW);
    float*  sV   = (float*) (smem + L_SV);
    float*  sRs  = (float*) (smem + L_SRS);
    float*  sStop= (float*) (smem + L_SSTOP);

    const int tid  = threadIdx.x;
    const int lane = tid & 31;
    const int wave = tid >> 5;
    const int bg   = blockIdx.x * 8;    // batch base of this group (8 rows)

    // ---- init recurrent state / cached constants ----
    for (int idx = tid; idx < 8 * 256; idx += 512) {
        int j = idx & 255;
        sHa[idx]  = A.attn_init[j];
        sH1[idx]  = A.dec_init[j];
        sH2[idx]  = A.dec_init[256 + j];
        sCtx[idx] = 0.0f;
    }
    for (int idx = tid; idx < 8 * 512; idx += 512) {
        sAt[idx] = 0.0f; sCu[idx] = 0.0f;
        int b = idx >> 9, tt = idx & 511;
        sMask[idx] = A.mask[(size_t)(bg + b) * 512 + tt];
    }
    for (int idx = tid; idx < 656; idx += 512) sSW[idx] = A.stop_W[idx];
    if (tid < 128) sV[tid] = A.v_W[tid];
    __syncthreads();

    for (int t = 0; t < 200; ++t) {
        // ---- P0: mem_in (= prev mel frame, or mem_init) -> sX, Kp=416
        for (int idx = tid; idx < 16 * 416; idx += 512) {
            int m = idx / 416, k = idx - m * 416;
            float v = 0.0f;
            if (m < 8 && k < 400)
                v = (t == 0) ? A.mem_init[k]
                             : A.memory[(size_t)(bg + m) * 80000 + (size_t)(t - 1) * 400 + k];
            sX[m * 416 + k] = (half_t)v;
        }
        // prefetch next step's teacher-forcing frame (13 x 128B lines per row)
        if (t < 199) {
            for (int idx = tid; idx < 8 * 13; idx += 512) {
                int b = idx / 13, j = idx - b * 13;
                __builtin_prefetch(A.memory + (size_t)(bg + b) * 80000 +
                                   (size_t)t * 400 + j * 32, 0, 0);
            }
        }
        __syncthreads();
        // ---- P1: prenet1 (400->256) + relu
        gemm16(sX, 416, A.preW1s, 16, 13, A.pre_b1, sG0, 256, lane, wave);
        __syncthreads();
        for (int idx = tid; idx < 16 * 256; idx += 512)
            sX[idx] = (half_t)((idx >> 8) < 8 ? fmaxf(sG0[idx], 0.0f) : 0.0f);
        __syncthreads();
        // ---- P2: prenet2 (256->128) + relu, then arnn input [x|ctx] Kp=384
        gemm16(sX, 256, A.preW2s, 8, 8, A.pre_b2, sG0, 128, lane, wave);
        __syncthreads();
        for (int idx = tid; idx < 16 * 384; idx += 512) {
            int m = idx / 384, j = idx - m * 384;
            float v = 0.0f;
            if (m < 8)
                v = (j < 128) ? fmaxf(sG0[m * 128 + j], 0.0f)
                              : sCtx[m * 256 + (j - 128)];
            sX[m * 384 + j] = (half_t)v;
        }
        __syncthreads();
        // ---- P3: attention GRU
        gemm16(sX, 384, A.aWihs, 48, 12, A.arnn_bih, sG0, 768, lane, wave);
        __syncthreads();
        for (int idx = tid; idx < 16 * 256; idx += 512)
            sX[idx] = (half_t)((idx >> 8) < 8 ? sHa[idx] : 0.0f);
        __syncthreads();
        gemm16(sX, 256, A.aWhhs, 48, 8, A.arnn_bhh, sG1, 768, lane, wave);
        __syncthreads();
        for (int idx = tid; idx < 8 * 256; idx += 512) {
            int b = idx >> 8, j = idx & 255;
            const float* gi = sG0 + b * 768; const float* gh = sG1 + b * 768;
            float r = fsigm(gi[j] + gh[j]);
            float z = fsigm(gi[256 + j] + gh[256 + j]);
            float n = ftanh(gi[512 + j] + r * gh[512 + j]);
            float h = (1.0f - z) * n + z * sHa[idx];
            sHa[idx] = h; sX[idx] = (half_t)h;
        }
        __syncthreads();
        // ---- P4: proc_q (256->128); zero energy accumulator
        gemm16(sX, 256, A.qWs, 8, 8, A.q_b, sAq, 128, lane, wave);
        for (int idx = tid; idx < 8 * 512; idx += 512) sE[idx] = 0.0f;
        __syncthreads();
        // ---- P5: location attention, T chunked by 64 (8 chunks of 512 rows)
        for (int c = 0; c < 8; ++c) {
            const int t0 = c * 64;
            // prefetch next chunk's proc_annot rows (2 x 128B per row)
            if (c < 7) {
                for (int idx = tid; idx < 1024; idx += 512) {
                    int b = idx >> 7, rem = idx & 127;
                    int tt = rem >> 1, hf = rem & 1;
                    __builtin_prefetch(A.annot +
                        ((size_t)(bg + b) * 512 + t0 + 64 + tt) * 128 + hf * 64, 0, 0);
                }
            }
            // im2col of [attn; attn_cum] 31-tap window -> sIm (512 x 64), LDS src
            for (int idx = tid; idx < 512 * 64; idx += 512) {
                int row = idx >> 6, k = idx & 63;
                int b = row >> 6, tt = row & 63;
                float v = 0.0f;
                if (k < 62) {
                    int c2 = (k < 31) ? 0 : 1;
                    int dt = k - c2 * 31;
                    int ts = t0 + tt + dt - 15;
                    if (ts >= 0 && ts < 512)
                        v = (c2 ? sCu : sAt)[b * 512 + ts];
                }
                sIm[idx] = (half_t)v;
            }
            __syncthreads();
            // conv as GEMM: (512 x 64) * (64 x 32) -> sLoc f16
            for (int tile = wave; tile < 64; tile += 16) {
                int mt = tile >> 1, nt = tile & 1;
                v8f acc = wmma_tile(sIm, 64, mt * 16, A.convS, 2, nt, 2, 0.0f, lane);
                int mAdd = (lane & 16) ? 8 : 0;
                int n = nt * 16 + (lane & 15);
#pragma unroll
                for (int r = 0; r < 8; ++r)
                    sLoc[(mt * 16 + r + mAdd) * 32 + n] = (half_t)acc[r];
            }
            __syncthreads();
            // loc proj (32->128) fused with tanh-energy dot into sE
            for (int tile = wave; tile < 256; tile += 16) {
                int mt = tile >> 3, nt = tile & 7;
                int n = nt * 16 + (lane & 15);
                v8f acc = wmma_tile(sLoc, 32, mt * 16, A.locWs, 8, nt, 1,
                                    A.loc_b[n], lane);
                int mAdd = (lane & 16) ? 8 : 0;
                float vn = sV[n];
#pragma unroll
                for (int r = 0; r < 8; ++r) {
                    int row = mt * 16 + r + mAdd;
                    int b = row >> 6, tt = row & 63;
                    int ts = t0 + tt;
                    float q  = sAq[b * 128 + n];
                    float an = (float)A.annot[((size_t)(bg + b) * 512 + ts) * 128 + n];
                    atomicAdd(&sE[b * 512 + ts], vn * ftanh(q + an + acc[r]));
                }
            }
            __syncthreads();
        }
        // ---- mask + sigmoid
        for (int idx = tid; idx < 8 * 512; idx += 512) {
            float e = sMask[idx] ? sE[idx] : -1e9f;
            sE[idx] = fsigm(e);
        }
        __syncthreads();
        // ---- per-row sum (wave b reduces row b)
        if (wave < 8) {
            float s = 0.0f;
            for (int j = lane; j < 512; j += 32) s += sE[wave * 512 + j];
#pragma unroll
            for (int off = 16; off; off >>= 1) s += __shfl_xor(s, off, 32);
            if (lane == 0) sRs[wave] = s;
        }
        __syncthreads();
        // ---- normalize -> align; update attn / attn_cum; emit alignment
        for (int idx = tid; idx < 8 * 512; idx += 512) {
            int b = idx >> 9, tt = idx & 511;
            float al = sE[idx] / sRs[b];
            sE[idx] = al;
            sAt[idx] = al;
            sCu[idx] += al;
            A.outAlign[((size_t)(bg + b) * 200 + t) * 512 + tt] = al;
        }
        __syncthreads();
        // ---- context: ctx[b,d] = sum_t align[b,t] * inputs[b,t,d]
        for (int idx = tid; idx < 8 * 256; idx += 512) {
            int b = idx >> 8, d = idx & 255;
            const float* inp = A.inputs + ((size_t)(bg + b) * 512) * 256 + d;
            const float* al  = sE + b * 512;
            float s = 0.0f;
#pragma unroll 8
            for (int tt = 0; tt < 512; ++tt) s += al[tt] * inp[(size_t)tt * 256];
            sCtx[idx] = s;
        }
        __syncthreads();
        // ---- P6: proj of [h_a | ctx] (512->256) -> dec_in (sD)
        for (int idx = tid; idx < 16 * 512; idx += 512) {
            int m = idx >> 9, j = idx & 511;
            float v = 0.0f;
            if (m < 8)
                v = (j < 256) ? sHa[m * 256 + j] : sCtx[m * 256 + (j - 256)];
            sX[idx] = (half_t)v;
        }
        __syncthreads();
        gemm16(sX, 512, A.projWs, 16, 16, A.proj_b, sD, 256, lane, wave);
        __syncthreads();
        // ---- P7: decoder GRU 1 + residual
        for (int idx = tid; idx < 16 * 256; idx += 512)
            sX[idx] = (half_t)((idx >> 8) < 8 ? sD[idx] : 0.0f);
        __syncthreads();
        gemm16(sX, 256, A.d1ihs, 48, 8, A.d1_bih, sG0, 768, lane, wave);
        __syncthreads();
        for (int idx = tid; idx < 16 * 256; idx += 512)
            sX[idx] = (half_t)((idx >> 8) < 8 ? sH1[idx] : 0.0f);
        __syncthreads();
        gemm16(sX, 256, A.d1hhs, 48, 8, A.d1_bhh, sG1, 768, lane, wave);
        __syncthreads();
        for (int idx = tid; idx < 8 * 256; idx += 512) {
            int b = idx >> 8, j = idx & 255;
            const float* gi = sG0 + b * 768; const float* gh = sG1 + b * 768;
            float r = fsigm(gi[j] + gh[j]);
            float z = fsigm(gi[256 + j] + gh[256 + j]);
            float n = ftanh(gi[512 + j] + r * gh[512 + j]);
            float h = (1.0f - z) * n + z * sH1[idx];
            sH1[idx] = h;
            float dv = h + sD[idx];
            sD[idx] = dv; sX[idx] = (half_t)dv;
        }
        __syncthreads();
        // ---- P8: decoder GRU 2 + residual
        gemm16(sX, 256, A.d2ihs, 48, 8, A.d2_bih, sG0, 768, lane, wave);
        __syncthreads();
        for (int idx = tid; idx < 16 * 256; idx += 512)
            sX[idx] = (half_t)((idx >> 8) < 8 ? sH2[idx] : 0.0f);
        __syncthreads();
        gemm16(sX, 256, A.d2hhs, 48, 8, A.d2_bhh, sG1, 768, lane, wave);
        __syncthreads();
        for (int idx = tid; idx < 8 * 256; idx += 512) {
            int b = idx >> 8, j = idx & 255;
            const float* gi = sG0 + b * 768; const float* gh = sG1 + b * 768;
            float r = fsigm(gi[j] + gh[j]);
            float z = fsigm(gi[256 + j] + gh[256 + j]);
            float n = ftanh(gi[512 + j] + r * gh[512 + j]);
            float h = (1.0f - z) * n + z * sH2[idx];
            sH2[idx] = h;
            float dv = h + sD[idx];        // dec_out
            sD[idx] = dv; sX[idx] = (half_t)dv;
        }
        __syncthreads();
        // ---- P9: mel head (256->400) + sigmoid; stop partials via LDS atomics
        if (tid < 16) sStop[tid] = 0.0f;
        __syncthreads();
        for (int nt = wave; nt < 25; nt += 16) {
            int n = nt * 16 + (lane & 15);
            v8f acc = wmma_tile(sX, 256, 0, A.melWs, 25, nt, 8, A.mel_b[n], lane);
            int mAdd = (lane & 16) ? 8 : 0;
            float sw = sSW[256 + n];
#pragma unroll
            for (int r = 0; r < 8; ++r) {
                int m = r + mAdd;
                if (m < 8) {
                    float o = fsigm(acc[r]);
                    A.outMel[((size_t)(bg + m) * 200 + t) * 400 + n] = o;
                    atomicAdd(&sStop[m], sw * o);
                }
            }
        }
        __syncthreads();
        // ---- P10: stop = stop_b + stop_W[0:256].dec_out + stop_W[256:].mel
        if (wave < 8) {
            float s = 0.0f;
            for (int d = lane; d < 256; d += 32) s += sSW[d] * sD[wave * 256 + d];
#pragma unroll
            for (int off = 16; off; off >>= 1) s += __shfl_xor(s, off, 32);
            if (lane == 0)
                A.outStop[(size_t)(bg + wave) * 200 + t] = s + sStop[wave] + A.stop_b[0];
        }
        __syncthreads();
    }
}

// ---------------------------------------------------------------------------
// Host side
// ---------------------------------------------------------------------------
static inline size_t alignUp(size_t x, size_t a) { return (x + a - 1) & ~(a - 1); }

extern "C" void kernel_launch(void* const* d_in, const int* in_sizes, int n_in,
                              void* d_out, int out_size, void* d_ws, size_t ws_size,
                              hipStream_t stream)
{
    (void)in_sizes; (void)n_in; (void)out_size; (void)ws_size;
    const float* inputs = (const float*)d_in[0];
    const float* memory = (const float*)d_in[1];
    const unsigned char* mask = (const unsigned char*)d_in[2];
    const float* pre_W1   = (const float*)d_in[3];
    const float* pre_b1   = (const float*)d_in[4];
    const float* pre_W2   = (const float*)d_in[5];
    const float* pre_b2   = (const float*)d_in[6];
    const float* arnn_Wih = (const float*)d_in[7];
    const float* arnn_Whh = (const float*)d_in[8];
    const float* arnn_bih = (const float*)d_in[9];
    const float* arnn_bhh = (const float*)d_in[10];
    const float* loc_convW= (const float*)d_in[11];
    const float* loc_W    = (const float*)d_in[12];
    const float* loc_b    = (const float*)d_in[13];
    const float* q_W      = (const float*)d_in[14];
    const float* q_b      = (const float*)d_in[15];
    const float* a_W      = (const float*)d_in[16];
    const float* a_b      = (const float*)d_in[17];
    const float* v_W      = (const float*)d_in[18];
    const float* proj_W   = (const float*)d_in[19];
    const float* proj_b   = (const float*)d_in[20];
    const float* d1_Wih   = (const float*)d_in[21];
    const float* d1_Whh   = (const float*)d_in[22];
    const float* d1_bih   = (const float*)d_in[23];
    const float* d1_bhh   = (const float*)d_in[24];
    const float* d2_Wih   = (const float*)d_in[25];
    const float* d2_Whh   = (const float*)d_in[26];
    const float* d2_bih   = (const float*)d_in[27];
    const float* d2_bhh   = (const float*)d_in[28];
    const float* mel_W    = (const float*)d_in[29];
    const float* mel_b    = (const float*)d_in[30];
    const float* stop_W   = (const float*)d_in[31];
    const float* stop_b   = (const float*)d_in[32];
    const float* attn_init= (const float*)d_in[33];
    const float* mem_init = (const float*)d_in[34];
    const float* dec_init = (const float*)d_in[35];

    char* ws = (char*)d_ws;
    size_t off = 0;
    half_t* annot = (half_t*)(ws + off); off += (size_t)64 * 512 * 128 * 2;   // 8 MB

    struct Swz { const float* W; int K, N; half_t* dst; };
    Swz tbl[14] = {
        { pre_W1,   400, 256, nullptr }, { pre_W2,   256, 128, nullptr },
        { arnn_Wih, 384, 768, nullptr }, { arnn_Whh, 256, 768, nullptr },
        { q_W,      256, 128, nullptr }, { loc_W,     32, 128, nullptr },
        { loc_convW, 62,  32, nullptr }, { a_W,      256, 128, nullptr },
        { proj_W,   512, 256, nullptr }, { d1_Wih,   256, 768, nullptr },
        { d1_Whh,   256, 768, nullptr }, { d2_Wih,   256, 768, nullptr },
        { d2_Whh,   256, 768, nullptr }, { mel_W,    256, 400, nullptr },
    };
    for (int i = 0; i < 14; ++i) {
        int Kp = (tbl[i].K + 31) & ~31, Np = (tbl[i].N + 15) & ~15;
        off = alignUp(off, 256);
        tbl[i].dst = (half_t*)(ws + off);
        off += (size_t)Kp * Np * 2;
        int total = Kp * Np;
        swz_kernel<<<(total + 255) / 256, 256, 0, stream>>>(
            tbl[i].W, tbl[i].dst, tbl[i].K, tbl[i].N, Kp, Np);
    }

    // precompute proc_annot with WMMA
    annot_kernel<<<2048, 256, 0, stream>>>(inputs, tbl[7].dst, a_b, annot);

    MainArgs A;
    A.inputs = inputs; A.memory = memory; A.mask = mask;
    A.pre_b1 = pre_b1; A.pre_b2 = pre_b2;
    A.arnn_bih = arnn_bih; A.arnn_bhh = arnn_bhh;
    A.loc_b = loc_b; A.q_b = q_b; A.v_W = v_W;
    A.proj_b = proj_b;
    A.d1_bih = d1_bih; A.d1_bhh = d1_bhh; A.d2_bih = d2_bih; A.d2_bhh = d2_bhh;
    A.mel_b = mel_b; A.stop_W = stop_W; A.stop_b = stop_b;
    A.attn_init = attn_init; A.mem_init = mem_init; A.dec_init = dec_init;
    A.annot = annot;
    A.preW1s = tbl[0].dst; A.preW2s = tbl[1].dst;
    A.aWihs = tbl[2].dst;  A.aWhhs = tbl[3].dst;
    A.qWs = tbl[4].dst;    A.locWs = tbl[5].dst;  A.convS = tbl[6].dst;
    A.projWs = tbl[8].dst;
    A.d1ihs = tbl[9].dst;  A.d1hhs = tbl[10].dst;
    A.d2ihs = tbl[11].dst; A.d2hhs = tbl[12].dst; A.melWs = tbl[13].dst;

    float* out = (float*)d_out;
    A.outMel   = out;                              // 64*200*400
    A.outAlign = out + (size_t)64 * 200 * 400;     // 64*200*512
    A.outStop  = out + (size_t)64 * 200 * 400 + (size_t)64 * 200 * 512;

    decoder_main<<<8, 512, LDS_BYTES, stream>>>(A);
}
